// Model_22909355557267
// MI455X (gfx1250) — matrix operand order repaired and checked by reading, not compiled, not run on hardware
//
#include <hip/hip_runtime.h>
#include <math.h>

typedef __attribute__((ext_vector_type(16))) _Float16 v16h;
typedef __attribute__((ext_vector_type(8)))  float    v8f;

#define TB 256

// ------------------------------------------------------------------
// WMMA GEMM:  C[N,M] = act( A[N,K] @ W[K,M] + bias )  (optionally += C)
// A pre-converted to f16 row-major; W pre-converted to f16 TRANSPOSED (M x K).
// v_wmma_f32_16x16x32_f16 with f32 accum.
// Block: 256 thr = 8 waves; block tile 128x64; wave tile 32x32 (2x2 accs).
// Fast path: double-buffered LDS fed by global_load_async_to_lds_b128;
// both A and B fragments are contiguous 16B runs in LDS -> ds_load_b128.
// ------------------------------------------------------------------
#define BM 128
#define BN 64
#define BK 32
#define LDA  40  // halves: rows 80B (16B aligned), bank stride 20 dw (conflict-free)
#define LDBT 40  // halves per B column-row (n-major), same properties

__global__ __launch_bounds__(TB)
void k_gemm(const _Float16* __restrict__ A, const _Float16* __restrict__ WT,
            const float* __restrict__ bias, float* __restrict__ C,
            int N, int K, int M, int act, int accumulate)
{
    __shared__ __align__(16) _Float16 As[2][BM * LDA];
    __shared__ __align__(16) _Float16 Bs[2][BN * LDBT];

    const int tid  = threadIdx.x;
    const int lane = tid & 31;
    const int wid  = tid >> 5;
    const int wm   = wid & 3;   // 4 M-groups of 32 rows
    const int wn   = wid >> 2;  // 2 N-groups of 32 cols
    const int lm   = lane & 15;
    const int hi   = lane >> 4;

    const int blockM = blockIdx.x * BM;
    const int blockN = blockIdx.y * BN;

    v8f acc[2][2] = {};

    // fragment load + 4x WMMA on one LDS buffer (documented wave32 layouts)
    auto compute_tile = [&](int b) {
        const _Float16* as = As[b];
        const _Float16* bs = Bs[b];
        v16h af[2], bf[2];
#pragma unroll
        for (int mi = 0; mi < 2; ++mi) {
            const _Float16* p = as + (wm * 32 + mi * 16 + lm) * LDA + 8 * hi;
#pragma unroll
            for (int h = 0; h < 8; ++h) {     // contiguous: k = 8*hi + h
                af[mi][h]     = p[h];
                af[mi][8 + h] = p[16 + h];    // contiguous: k = 16 + 8*hi + h
            }
        }
#pragma unroll
        for (int ni = 0; ni < 2; ++ni) {
            const _Float16* p = bs + (wn * 32 + ni * 16 + lm) * LDBT + 8 * hi;
#pragma unroll
            for (int h = 0; h < 16; ++h)      // k = 8*hi + h (h<8), 16 + 8*hi + (h-8) (h>=8)
                bf[ni][h] = p[h < 8 ? h : h + 8];
        }
#pragma unroll
        for (int mi = 0; mi < 2; ++mi)
#pragma unroll
            for (int ni = 0; ni < 2; ++ni)
                acc[mi][ni] = __builtin_amdgcn_wmma_f32_16x16x32_f16(
                    false, af[mi], false, bf[ni],
                    (short)0, acc[mi][ni], false, false);
    };

    const bool fast = (blockM + BM <= N) && (blockN + BN <= M) && ((K & (BK - 1)) == 0);

    if (fast) {
        // async DMA: A tile 128x32 (2 x b128/thr) + B tile 64colsx32k (1 x b128/thr)
        auto issue_tile = [&](int t, int b) {
            const int kb = t * BK;
#pragma unroll
            for (int i = 0; i < 2; ++i) {
                int e = tid + i * 256;       // 512 x 16B ops
                int r = e >> 2;              // 4 ops per 32-half row
                int c = (e & 3) * 8;
                unsigned long long ga =
                    (unsigned long long)(A + (long long)(blockM + r) * K + kb + c);
                unsigned ld = (unsigned)(uintptr_t)&As[b][r * LDA + c];
                asm volatile("global_load_async_to_lds_b128 %0, %1, off"
                             :: "v"(ld), "v"(ga) : "memory");
            }
            {
                int n = tid >> 2;            // 4 ops per 32-half column
                int c = (tid & 3) * 8;
                unsigned long long ga =
                    (unsigned long long)(WT + (long long)(blockN + n) * K + kb + c);
                unsigned ld = (unsigned)(uintptr_t)&Bs[b][n * LDBT + c];
                asm volatile("global_load_async_to_lds_b128 %0, %1, off"
                             :: "v"(ld), "v"(ga) : "memory");
            }
        };

        const int T = K / BK;
        issue_tile(0, 0);
        for (int t = 0; t < T; ++t) {
            if (t + 1 < T) {
                issue_tile(t + 1, (t + 1) & 1);
                asm volatile("s_wait_asynccnt 0x3" ::: "memory"); // tile t landed (in-order)
            } else {
                asm volatile("s_wait_asynccnt 0x0" ::: "memory");
            }
            __syncthreads();
            compute_tile(t & 1);
            __syncthreads();
        }
    } else {
        // guarded edge path (last row block / M=4 decoder / K=8 encoder)
        for (int kb = 0; kb < K; kb += BK) {
#pragma unroll
            for (int i = 0; i < 16; ++i) {
                int e = tid + i * 256;
                int r = e >> 5, k = e & 31;
                int gr = blockM + r, gk = kb + k;
                As[0][r * LDA + k] = (gr < N && gk < K)
                    ? A[(long long)gr * K + gk] : (_Float16)0.f;
            }
#pragma unroll
            for (int i = 0; i < 8; ++i) {
                int e = tid + i * 256;       // 64 cols x 32 k
                int n = e >> 5, k = e & 31;
                int gn = blockN + n, gk = kb + k;
                Bs[0][n * LDBT + k] = (gn < M && gk < K)
                    ? WT[(long long)gn * K + gk] : (_Float16)0.f;
            }
            __syncthreads();
            compute_tile(0);
            __syncthreads();
        }
    }

    // epilogue: C/D layout -> m = r + 8*hi, n = lane&15
#pragma unroll
    for (int mi = 0; mi < 2; ++mi) {
#pragma unroll
        for (int ni = 0; ni < 2; ++ni) {
            int col  = blockN + wn * 32 + ni * 16 + lm;
            int rowb = blockM + wm * 32 + mi * 16 + 8 * hi;
            if (col < M) {
                float bv = bias ? bias[col] : 0.f;
#pragma unroll
                for (int r = 0; r < 8; ++r) {
                    int row = rowb + r;
                    if (row < N) {
                        float v = acc[mi][ni][r] + bv;
                        if (act == 1)      v = fmaxf(v, 0.f);
                        else if (act == 2) v = 0.5f * v * (1.f + erff(v * 0.70710678118f));
                        long long o = (long long)row * M + col;
                        if (accumulate) v += C[o];
                        C[o] = v;
                    }
                }
            }
        }
    }
}

// ------------------------- elementwise / graph kernels -------------------------
__global__ void k_cvt_f16(const float* __restrict__ x, _Float16* __restrict__ y, long long n) {
    long long i = (long long)blockIdx.x * blockDim.x + threadIdx.x;
    if (i < n) y[i] = (_Float16)x[i];
}

// y[m*K + k] = (f16) x[k*M + m]   (weight convert + transpose)
__global__ void k_cvt_tr(const float* __restrict__ x, _Float16* __restrict__ y, int K, int M) {
    long long i = (long long)blockIdx.x * blockDim.x + threadIdx.x;
    if (i >= (long long)K * M) return;
    int m = (int)(i % M);
    long long k = i / M;
    y[(long long)m * K + k] = (_Float16)x[k * M + m];
}

__global__ void k_zero(float* __restrict__ p, long long n) {
    long long i = (long long)blockIdx.x * blockDim.x + threadIdx.x;
    if (i < n) p[i] = 0.f;
}

__global__ void k_count(const int* __restrict__ dst, float* __restrict__ cnt, long long E) {
    long long i = (long long)blockIdx.x * blockDim.x + threadIdx.x;
    if (i < E) atomicAdd(&cnt[dst[i]], 1.f);
}

__global__ void k_scatter_add(const float* __restrict__ x, const int* __restrict__ src,
                              const int* __restrict__ dst, float* __restrict__ acc,
                              long long total, int C) {
    long long i = (long long)blockIdx.x * blockDim.x + threadIdx.x;
    if (i >= total) return;
    long long e = i / C;
    int c = (int)(i % C);
    atomicAdd(&acc[(long long)dst[e] * C + c], x[(long long)src[e] * C + c]);
}

__global__ void k_div_cnt(float* __restrict__ acc, const float* __restrict__ cnt,
                          long long total, int C) {
    long long i = (long long)blockIdx.x * blockDim.x + threadIdx.x;
    if (i >= total) return;
    acc[i] /= fmaxf(cnt[i / C], 1.f);
}

// y[r, off:off+Cx] = x[idx ? idx[r] : r, :]   (y has Cy cols)
__global__ void k_gather_cols(const float* __restrict__ x, const int* __restrict__ idx,
                              float* __restrict__ y, long long Nout, int Cx, int Cy, int off) {
    long long i = (long long)blockIdx.x * blockDim.x + threadIdx.x;
    if (i >= Nout * (long long)Cx) return;
    long long r = i / Cx;
    int c = (int)(i % Cx);
    long long sr = idx ? (long long)idx[r] : r;
    y[r * Cy + off + c] = x[sr * Cx + c];
}

__global__ __launch_bounds__(TB)
void k_bn_stats(const float* __restrict__ x, float* __restrict__ mean,
                float* __restrict__ var, int N, int C) {
    int c = blockIdx.x;
    __shared__ float s1[TB], s2[TB];
    float a = 0.f, b = 0.f;
    for (int r = threadIdx.x; r < N; r += TB) {
        float v = x[(long long)r * C + c];
        a += v; b += v * v;
    }
    s1[threadIdx.x] = a; s2[threadIdx.x] = b;
    __syncthreads();
    for (int s = TB / 2; s > 0; s >>= 1) {
        if (threadIdx.x < s) { s1[threadIdx.x] += s1[threadIdx.x + s];
                               s2[threadIdx.x] += s2[threadIdx.x + s]; }
        __syncthreads();
    }
    if (threadIdx.x == 0) {
        float m = s1[0] / (float)N;
        mean[c] = m;
        var[c]  = s2[0] / (float)N - m * m;   // biased var (jnp.var)
    }
}

__global__ void k_bn_apply(const float* __restrict__ x, const float* __restrict__ mean,
                           const float* __restrict__ var, const float* __restrict__ g,
                           const float* __restrict__ b, float* __restrict__ y,
                           long long total, int C, int relu) {
    long long i = (long long)blockIdx.x * blockDim.x + threadIdx.x;
    if (i >= total) return;
    int c = (int)(i % C);
    float v = (x[i] - mean[c]) * rsqrtf(var[c] + 1e-5f) * g[c] + b[c];
    if (relu) v = fmaxf(v, 0.f);
    y[i] = v;
}

// ------------------------------------------------------------------
extern "C" void kernel_launch(void* const* d_in, const int* in_sizes, int n_in,
                              void* d_out, int out_size, void* d_ws, size_t ws_size,
                              hipStream_t stream)
{
    (void)n_in; (void)out_size; (void)ws_size;
    const float* fx  = (const float*)d_in[0];
    const int*   geo = (const int*)d_in[1];
    const int* ids[4] = {(const int*)d_in[2], (const int*)d_in[3],
                         (const int*)d_in[4], (const int*)d_in[5]};
    const int* eis[4] = {(const int*)d_in[6], (const int*)d_in[7],
                         (const int*)d_in[8], (const int*)d_in[9]};
    const int* cls[4] = {(const int*)d_in[10], (const int*)d_in[11],
                         (const int*)d_in[12], (const int*)d_in[13]};
    auto F = [&](int i) { return (const float*)d_in[i]; };
    // params flattened in setup_inputs() insertion order:
    // 14 enc_w1, 15 enc_b1, 16 enc_w2, 17 enc_b2
    // 18..32 down[i]{wl,bl,wr}, 33..44 up[i]{wl,bl,wr}
    // 45..62 bn[i]{g,b}, 63 dec_w1, 64 dec_b1, 65 dec_w2, 66 dec_b2

    int Nl[5] = { in_sizes[0] / 8, in_sizes[2], in_sizes[3], in_sizes[4], in_sizes[5] };
    int Cc[5] = { 128, 256, 512, 1024, 2048 };
    long long E0 = in_sizes[1] / 2;
    long long Ee[4] = { in_sizes[6] / 2, in_sizes[7] / 2, in_sizes[8] / 2, in_sizes[9] / 2 };

    float* ws = (float*)d_ws;
    long long off = 0;
    float* AGG = ws + off; off += 19200000;   // max conv input (N3*3072 / N0*384)
    float* CAT = ws + off; off += 19200000;   // gathered / concat conv input
    float* TMP = ws + off; off += 12800000;   // conv out / MLP hidden (N0*256)
    float* Z[5];
    for (int i = 0; i < 5; ++i) { Z[i] = ws + off; off += 6400000; }
    float* ZE   = ws + off; off += 6400000;
    float* CNT  = ws + off; off += 50048;
    float* MEAN = ws + off; off += 2048;
    float* VARV = ws + off; off += 2048;
    _Float16* A16 = (_Float16*)(ws + off); off += 9600000;   // 19.2e6 halves
    _Float16* W16 = (_Float16*)(ws + off); off += 1600000;   // 3.2e6 halves (W^T)

    auto cdiv = [](long long a, long long b) { return (a + b - 1) / b; };
    auto cvt = [&](const float* x, _Float16* y, long long n) {
        k_cvt_f16<<<(unsigned)cdiv(n, TB), TB, 0, stream>>>(x, y, n);
    };
    auto gemm = [&](const float* A, const float* Wt, const float* b, float* Co,
                    int n, int k, int m, int act, int acc) {
        cvt(A, A16, (long long)n * k);
        k_cvt_tr<<<(unsigned)cdiv((long long)k * m, TB), TB, 0, stream>>>(Wt, W16, k, m);
        dim3 g((unsigned)cdiv(n, BM), (unsigned)cdiv(m, BN));
        k_gemm<<<g, TB, 0, stream>>>(A16, W16, b, Co, n, k, m, act, acc);
    };
    auto zero = [&](float* p, long long n) {
        k_zero<<<(unsigned)cdiv(n, TB), TB, 0, stream>>>(p, n);
    };
    // SAGE: out = mean_agg(x)@wl + bl + x@wr
    auto sage = [&](const float* x, int n, int cin, const int* ei, long long E,
                    const float* wl, const float* bl, const float* wr,
                    float* out, int cout) {
        const int* src = ei;
        const int* dst = ei + E;
        zero(AGG, (long long)n * cin);
        zero(CNT, n);
        k_count<<<(unsigned)cdiv(E, TB), TB, 0, stream>>>(dst, CNT, E);
        k_scatter_add<<<(unsigned)cdiv(E * cin, TB), TB, 0, stream>>>(x, src, dst, AGG, E * cin, cin);
        k_div_cnt<<<(unsigned)cdiv((long long)n * cin, TB), TB, 0, stream>>>(AGG, CNT, (long long)n * cin, cin);
        gemm(AGG, wl, bl, out, n, cin, cout, 0, 0);
        gemm(x,  wr, nullptr, out, n, cin, cout, 0, 1);
    };
    auto bnact = [&](const float* x, float* y, int n, int c,
                     const float* g, const float* b, int relu) {
        k_bn_stats<<<c, TB, 0, stream>>>(x, MEAN, VARV, n, c);
        k_bn_apply<<<(unsigned)cdiv((long long)n * c, TB), TB, 0, stream>>>(
            x, MEAN, VARV, g, b, y, (long long)n * c, c, relu);
    };

    // ---------------- encoder MLP ----------------
    gemm(fx,  F(14), F(15), TMP, Nl[0], 8,   256, 2, 0);   // GELU
    gemm(TMP, F(16), F(17), ZE,  Nl[0], 256, 128, 0, 0);

    // ---------------- level-0 conv ----------------
    sage(ZE, Nl[0], 128, geo, E0, F(18), F(19), F(20), TMP, 128);
    bnact(TMP, Z[0], Nl[0], 128, F(45), F(46), 1);

    // ---------------- down path ----------------
    for (int l = 0; l < 4; ++l) {
        int cin = Cc[l], cout = Cc[l + 1], nn = Nl[l + 1];
        k_gather_cols<<<(unsigned)cdiv((long long)nn * cin, TB), TB, 0, stream>>>(
            Z[l], ids[l], CAT, nn, cin, cin, 0);
        sage(CAT, nn, cin, eis[l], Ee[l],
             F(21 + 3 * l), F(22 + 3 * l), F(23 + 3 * l), TMP, cout);
        bnact(TMP, Z[l + 1], nn, cout, F(47 + 2 * l), F(48 + 2 * l), 1);
    }

    // ---------------- up path ----------------
    for (int n = 4; n >= 1; --n) {
        int cz = Cc[n], cs = Cc[n - 1], cin = cz + cs, cout = Cc[n - 1];
        int nu = Nl[n - 1];
        // upsample current z then concat skip
        k_gather_cols<<<(unsigned)cdiv((long long)nu * cz, TB), TB, 0, stream>>>(
            Z[n], cls[n - 1], CAT, nu, cz, cin, 0);
        k_gather_cols<<<(unsigned)cdiv((long long)nu * cs, TB), TB, 0, stream>>>(
            Z[n - 1], nullptr, CAT, nu, cs, cin, cz);
        const int* ei = (n == 1) ? geo : eis[n - 2];
        long long  E  = (n == 1) ? E0  : Ee[n - 2];
        int ui = n - 1;
        sage(CAT, nu, cin, ei, E,
             F(33 + 3 * ui), F(34 + 3 * ui), F(35 + 3 * ui), TMP, cout);
        int bni = 4 + n;
        bnact(TMP, Z[n - 1], nu, cout, F(45 + 2 * bni), F(46 + 2 * bni), (n != 1) ? 1 : 0);
    }

    // ---------------- decoder MLP ----------------
    gemm(Z[0], F(63), F(64), TMP, Nl[0], 128, 256, 2, 0);          // GELU
    gemm(TMP,  F(65), F(66), (float*)d_out, Nl[0], 256, 4, 0, 0);  // [N0,4]
}
